// ProteinMultiScaleTransformer_39960375722284
// MI455X (gfx1250) — compile-verified, matrix operand order checked by reading, not run on hardware
//
#include <hip/hip_runtime.h>
#include <math.h>

#define B_ 2
#define L_ 1024
#define C_ 4
#define E_ 256
#define H_ 8
#define D_ 32
#define NL_ 2
#define WIN_ 256
#define NC_ 10
#define HID_ 512
#define IN_PAIR_ 1028   /* C*(WIN+1) = 4*257 */
#define KPAD_ 1056      /* IN_PAIR_ rounded up to multiple of 32 */
#define M_ (B_ * L_)    /* 2048 */

typedef unsigned int       u32;
typedef unsigned long long u64;
typedef __attribute__((ext_vector_type(16))) _Float16 v16h;
typedef __attribute__((ext_vector_type(8)))  _Float16 v8h;
typedef __attribute__((ext_vector_type(8)))  float    v8f;
typedef __attribute__((ext_vector_type(4)))  u32      u32x4;
typedef __attribute__((ext_vector_type(8)))  u32      u32x8;

#define WMMA_F16(a, b, c) \
  __builtin_amdgcn_wmma_f32_16x16x32_f16(false, (a), false, (b), (short)0, (c), false, false)

// ---------------------------------------------------------------------------
// Tensor Data Mover: DMA a rows x elems_w 2D tile (row stride row_stride
// elements) from global into LDS. Groups 0+1 only (<=2D, ISA ch 8).
// data_size code: 0=1B 1=2B 2=4B 3=8B.
// ---------------------------------------------------------------------------
__device__ __forceinline__ void tdm_load_2d(void* lds_dst, const void* gsrc,
                                            u32 dsz, u32 elems_w, u32 rows,
                                            u32 row_stride, u32 tensor_w) {
  u32 lds_off = (u32)(uintptr_t)lds_dst;          // low 32 bits = LDS offset
  u64 ga = (u64)(uintptr_t)gsrc;
  u32 d0 = __builtin_amdgcn_readfirstlane(1u);    // count=1, user mode
  u32 d1 = __builtin_amdgcn_readfirstlane(lds_off);
  u32 d2 = __builtin_amdgcn_readfirstlane((u32)ga);
  u32 d3 = __builtin_amdgcn_readfirstlane(((u32)(ga >> 32) & 0x1FFFFFFu) | (2u << 30));
  u32x4 g0 = {d0, d1, d2, d3};
  u32 e0 = __builtin_amdgcn_readfirstlane(dsz << 16);
  u32 e1 = __builtin_amdgcn_readfirstlane((tensor_w & 0xFFFFu) << 16);          // dim0 lo
  u32 e2 = __builtin_amdgcn_readfirstlane((tensor_w >> 16) | ((rows & 0xFFFFu) << 16)); // dim0 hi|dim1 lo
  u32 e3 = __builtin_amdgcn_readfirstlane((rows >> 16) | (elems_w << 16));      // dim1 hi|tile_dim0
  u32 e4 = __builtin_amdgcn_readfirstlane(rows & 0xFFFFu);                      // tile_dim1
  u32 e5 = __builtin_amdgcn_readfirstlane(row_stride);                          // dim0_stride lo32
  u32x8 g1 = {e0, e1, e2, e3, e4, e5, 0u, 0u};
  asm volatile("tensor_load_to_lds %0, %1" :: "s"(g0), "s"(g1) : "memory");
}

// ---------------------------------------------------------------------------
// WMMA tile loaders (wave32, 16-bit A 16x32 layout per CDNA5 ISA 7.12.2).
// ---------------------------------------------------------------------------
__device__ __forceinline__ v16h load_a_tile_f32(const float* __restrict__ A, int lda,
                                                int row0, int k0, int lane) {
  int m  = row0 + (lane & 15);
  int kk = k0 + ((lane >> 4) << 3);
  const float* p = A + (size_t)m * lda + kk;
  float4 x0 = *(const float4*)(p);
  float4 x1 = *(const float4*)(p + 4);
  float4 y0 = *(const float4*)(p + 16);
  float4 y1 = *(const float4*)(p + 20);
  v16h a;
  a[0] = (_Float16)x0.x;  a[1] = (_Float16)x0.y;  a[2]  = (_Float16)x0.z;  a[3]  = (_Float16)x0.w;
  a[4] = (_Float16)x1.x;  a[5] = (_Float16)x1.y;  a[6]  = (_Float16)x1.z;  a[7]  = (_Float16)x1.w;
  a[8] = (_Float16)y0.x;  a[9] = (_Float16)y0.y;  a[10] = (_Float16)y0.z;  a[11] = (_Float16)y0.w;
  a[12]= (_Float16)y1.x;  a[13]= (_Float16)y1.y;  a[14] = (_Float16)y1.z;  a[15] = (_Float16)y1.w;
  return a;
}

__device__ __forceinline__ v16h load_a_tile_f16(const _Float16* __restrict__ A, int lda,
                                                int row0, int k0, int lane) {
  int m  = row0 + (lane & 15);
  int kk = k0 + ((lane >> 4) << 3);
  const _Float16* p = A + (size_t)m * lda + kk;
  v8h lo = *(const v8h*)(p);
  v8h hi = *(const v8h*)(p + 16);
  v16h a;
#pragma unroll
  for (int i = 0; i < 8; ++i) { a[i] = lo[i]; a[i + 8] = hi[i]; }
  return a;
}

// B tile from a K x 32 f16 LDS panel; half = 0 or 16 selects the column group.
__device__ __forceinline__ v16h load_b_lds32(const _Float16* Bs, int k0, int half, int lane) {
  int n    = half + (lane & 15);
  int koff = (lane >> 4) << 3;
  const _Float16* p = Bs + (size_t)(k0 + koff) * 32 + n;
  v16h b;
#pragma unroll
  for (int i = 0; i < 8; ++i) {
    b[i]     = p[i * 32];
    b[i + 8] = p[(16 + i) * 32];
  }
  return b;
}

// B tile from transposed f16 V ([E][M] layout): contiguous along keys.
__device__ __forceinline__ v16h load_b_vt(const _Float16* __restrict__ VT, int Mtot,
                                          int col0, int kbase, int lane) {
  int n  = col0 + (lane & 15);
  int kk = kbase + ((lane >> 4) << 3);
  const _Float16* p = VT + (size_t)n * Mtot + kk;
  v8h lo = *(const v8h*)(p);
  v8h hi = *(const v8h*)(p + 16);
  v16h b;
#pragma unroll
  for (int i = 0; i < 8; ++i) { b[i] = lo[i]; b[i + 8] = hi[i]; }
  return b;
}

__device__ __forceinline__ float gelu_f(float v) {
  return 0.5f * v * (1.0f + erff(v * 0.70710678118654752440f));
}

// ---------------------------------------------------------------------------
// One-time (per launch) conversions.
// ---------------------------------------------------------------------------
__global__ __launch_bounds__(256)
void cvt_f16_kernel(const float* __restrict__ src, _Float16* __restrict__ dst, int n) {
  int i = blockIdx.x * 256 + threadIdx.x;
  if (i < n) dst[i] = (_Float16)src[i];
}

__global__ __launch_bounds__(256)
void cvt_w1_kernel(const float* __restrict__ src, _Float16* __restrict__ dst) {
  int i = blockIdx.x * 256 + threadIdx.x;
  if (i >= KPAD_ * E_) return;
  int row = i >> 8;
  dst[i] = (row < IN_PAIR_) ? (_Float16)src[i] : (_Float16)0.0f;
}

// flat[b,l, c*257+w] = x_pairwise[b,c,l, l+w-128]  (f16, zero padded)
__global__ __launch_bounds__(256)
void pair_flat_kernel(const float* __restrict__ xp, _Float16* __restrict__ flat) {
  int i = blockIdx.x * 256 + threadIdx.x;
  if (i >= M_ * KPAD_) return;
  int m = i / KPAD_;
  int k = i - m * KPAD_;
  int b = m >> 10;
  int l = m & (L_ - 1);
  float v = 0.0f;
  if (k < IN_PAIR_) {
    int c = k / 257;
    int w = k - c * 257;
    int pos = l + w - (WIN_ / 2);
    if (pos >= 0 && pos < L_)
      v = xp[((((size_t)b * C_ + c) * L_) + l) * L_ + pos];
  }
  flat[i] = (_Float16)v;
}

// ---------------------------------------------------------------------------
// GEMM epilogue for one 16x16 quadrant (8 rows per lane, fixed column).
// ---------------------------------------------------------------------------
__device__ __forceinline__ void gemm_store(v8f acc, int rbase, int col,
                                           const float* __restrict__ bias,
                                           const float* __restrict__ res,
                                           float* __restrict__ out, int ldo,
                                           _Float16* __restrict__ out_h,
                                           _Float16* __restrict__ out_vt,
                                           int act) {
  float bv = bias ? bias[col] : 0.0f;
  float vals[8];
#pragma unroll
  for (int r = 0; r < 8; ++r) {
    float v = acc[r] + bv;
    if (act == 1) v = fmaxf(v, 0.0f);
    else if (act == 2) v = gelu_f(v);
    if (res) v += res[(size_t)(rbase + r) * ldo + col];
    vals[r] = v;
  }
  if (out) {
#pragma unroll
    for (int r = 0; r < 8; ++r) out[(size_t)(rbase + r) * ldo + col] = vals[r];
  }
  if (out_h) {
#pragma unroll
    for (int r = 0; r < 8; ++r) out_h[(size_t)(rbase + r) * ldo + col] = (_Float16)vals[r];
  }
  if (out_vt) {  // [N][M_] layout; 8 consecutive rows -> one packed b128 store
    v8h pk;
#pragma unroll
    for (int r = 0; r < 8; ++r) pk[r] = (_Float16)vals[r];
    *(v8h*)(out_vt + (size_t)col * M_ + rbase) = pk;
  }
}

// ---------------------------------------------------------------------------
// Generic GEMM: 4 waves/block, 64 rows x 32 cols; B panel (K x 32 f16) via TDM.
// Optional outputs: f32, f16 row-major copy, f16 transposed copy.
// ---------------------------------------------------------------------------
#define GEMM_KMAX 512
__global__ __launch_bounds__(128)
void gemm_wmma_kernel(const float* __restrict__ A, int lda,
                      const _Float16* __restrict__ Wh, int ldw,
                      const float* __restrict__ bias,
                      const float* __restrict__ res,
                      float* __restrict__ out, int ldo,
                      _Float16* __restrict__ out_h,
                      _Float16* __restrict__ out_vt,
                      int K, int act) {
  __shared__ _Float16 Bs[GEMM_KMAX * 32];
  int lane = threadIdx.x & 31;
  int wave = threadIdx.x >> 5;
  int row0 = (blockIdx.x * 4 + wave) * 16;
  int col0 = blockIdx.y * 32;

  if (wave == 0) {
    tdm_load_2d(Bs, Wh + col0, /*f16*/1, 32, K, ldw, ldw);
    __builtin_amdgcn_s_wait_tensorcnt(0);
  }
  __syncthreads();

  v8f acc0 = {}, acc1 = {};
  for (int k0 = 0; k0 < K; k0 += 32) {
    v16h a  = load_a_tile_f32(A, lda, row0, k0, lane);
    v16h b0 = load_b_lds32(Bs, k0, 0,  lane);
    v16h b1 = load_b_lds32(Bs, k0, 16, lane);
    acc0 = WMMA_F16(a, b0, acc0);
    acc1 = WMMA_F16(a, b1, acc1);
  }

  int nloc  = lane & 15;
  int rbase = row0 + ((lane >> 4) << 3);
  gemm_store(acc0, rbase, col0 + nloc,      bias, res, out, ldo, out_h, out_vt, act);
  gemm_store(acc1, rbase, col0 + 16 + nloc, bias, res, out, ldo, out_h, out_vt, act);
}

// Pair MLP layer 1: ph = relu(flat_f16 @ w1_f16 + b1), K = 1056 (f16 A).
__global__ __launch_bounds__(128)
void pair_hid_kernel(const _Float16* __restrict__ flat, const _Float16* __restrict__ w1h,
                     const float* __restrict__ b1, float* __restrict__ ph) {
  __shared__ _Float16 Bs[KPAD_ * 32];
  int lane = threadIdx.x & 31;
  int wave = threadIdx.x >> 5;
  int row0 = (blockIdx.x * 4 + wave) * 16;
  int col0 = blockIdx.y * 32;

  if (wave == 0) {
    tdm_load_2d(Bs, w1h + col0, 1, 32, KPAD_, E_, E_);
    __builtin_amdgcn_s_wait_tensorcnt(0);
  }
  __syncthreads();

  v8f acc0 = {}, acc1 = {};
  for (int k0 = 0; k0 < KPAD_; k0 += 32) {
    v16h a  = load_a_tile_f16(flat, KPAD_, row0, k0, lane);
    v16h b0 = load_b_lds32(Bs, k0, 0,  lane);
    v16h b1 = load_b_lds32(Bs, k0, 16, lane);
    acc0 = WMMA_F16(a, b0, acc0);
    acc1 = WMMA_F16(a, b1, acc1);
  }

  int nloc  = lane & 15;
  int rbase = row0 + ((lane >> 4) << 3);
  gemm_store(acc0, rbase, col0 + nloc,      b1, nullptr, ph, E_, nullptr, nullptr, 1);
  gemm_store(acc1, rbase, col0 + 16 + nloc, b1, nullptr, ph, E_, nullptr, nullptr, 1);
}

// Pair MLP layer 2 + embeddings: x = ph @ w2 + b2 + token_emb[tok] + pe.
__global__ __launch_bounds__(128)
void pair_out_kernel(const float* __restrict__ ph, const _Float16* __restrict__ w2h,
                     const float* __restrict__ b2, const int* __restrict__ tokens,
                     const float* __restrict__ emb, const float* __restrict__ pe,
                     float* __restrict__ xout) {
  __shared__ _Float16 Bs[E_ * 32];
  int lane = threadIdx.x & 31;
  int wave = threadIdx.x >> 5;
  int row0 = (blockIdx.x * 4 + wave) * 16;
  int col0 = blockIdx.y * 32;

  if (wave == 0) {
    tdm_load_2d(Bs, w2h + col0, 1, 32, E_, E_, E_);
    __builtin_amdgcn_s_wait_tensorcnt(0);
  }
  __syncthreads();

  v8f acc0 = {}, acc1 = {};
  for (int k0 = 0; k0 < E_; k0 += 32) {
    v16h a  = load_a_tile_f32(ph, E_, row0, k0, lane);
    v16h b0 = load_b_lds32(Bs, k0, 0,  lane);
    v16h b1 = load_b_lds32(Bs, k0, 16, lane);
    acc0 = WMMA_F16(a, b0, acc0);
    acc1 = WMMA_F16(a, b1, acc1);
  }

  int nloc  = lane & 15;
  int rbase = row0 + ((lane >> 4) << 3);
#pragma unroll
  for (int half = 0; half < 2; ++half) {
    int col = col0 + half * 16 + nloc;
    v8f acc = half ? acc1 : acc0;
    float bv = b2[col];
#pragma unroll
    for (int r = 0; r < 8; ++r) {
      int row = rbase + r;
      int l = row & (L_ - 1);
      int tok = tokens[row];
      xout[(size_t)row * E_ + col] =
          acc[r] + bv + emb[(size_t)tok * E_ + col] + pe[(size_t)l * E_ + col];
    }
  }
}

// ---------------------------------------------------------------------------
// LayerNorm over E=256, one wave per (b,l) row.
// ---------------------------------------------------------------------------
__global__ __launch_bounds__(32)
void layernorm_kernel(const float* __restrict__ x, const float* __restrict__ g,
                      const float* __restrict__ bta, float* __restrict__ out) {
  int row = blockIdx.x;
  int lane = threadIdx.x;
  const float* xr = x + (size_t)row * E_;
  float vals[8];
  float s = 0.0f;
#pragma unroll
  for (int i = 0; i < 8; ++i) {
    vals[i] = xr[lane + 32 * i];
    s += vals[i];
  }
#pragma unroll
  for (int m = 16; m >= 1; m >>= 1) s += __shfl_xor(s, m, 32);
  float mean = s * (1.0f / (float)E_);
  float s2 = 0.0f;
#pragma unroll
  for (int i = 0; i < 8; ++i) {
    float d = vals[i] - mean;
    s2 += d * d;
  }
#pragma unroll
  for (int m = 16; m >= 1; m >>= 1) s2 += __shfl_xor(s2, m, 32);
  float rstd = rsqrtf(s2 * (1.0f / (float)E_) + 1e-5f);
#pragma unroll
  for (int i = 0; i < 8; ++i) {
    int c = lane + 32 * i;
    out[(size_t)row * E_ + c] = (vals[i] - mean) * rstd * g[c] + bta[c];
  }
}

// ---------------------------------------------------------------------------
// Fused biased attention, flash-style. One wave per (b, h, 16-query tile).
// Q/K: f16 row-major (b128 A loads). V: f16 transposed [E][M] (b128 B loads).
// Bias tiles (16x32 f32) double-buffered through LDS via TDM.
// ---------------------------------------------------------------------------
__global__ __launch_bounds__(32)
void attention_kernel(const _Float16* __restrict__ qh, const _Float16* __restrict__ kh,
                      const _Float16* __restrict__ vt, const float* __restrict__ bias,
                      const float* __restrict__ mask, const float* __restrict__ gate,
                      float* __restrict__ out, int layer) {
  __shared__ _Float16 pT[16][40];
  __shared__ float biasLds[2][16 * 32];
  int lane = threadIdx.x;
  int qt = blockIdx.x;
  int h  = blockIdx.y;
  int b  = blockIdx.z;
  int row0 = qt * 16;           // within batch
  int grow0 = b * L_ + row0;    // global row
  int nloc = lane & 15;
  int koff = (lane >> 4) << 3;
  const float scale = 0.17677669529663687f;  // D^-0.5
  float gv = gate[layer * H_ + h];

  const float* biasbh = bias + ((size_t)b * H_ + h) * L_ * L_;
  const float* mrow = mask + (size_t)b * L_;

  v16h qa = load_a_tile_f16(qh, E_, grow0, h * D_, lane);

  float mstate[8], lstate[8];
  v8f o0 = {}, o1 = {};
#pragma unroll
  for (int r = 0; r < 8; ++r) { mstate[r] = -1e30f; lstate[r] = 0.0f; }

  // prime the bias pipeline: tile for j0 = 0
  tdm_load_2d(biasLds[0], biasbh + (size_t)row0 * L_, /*f32*/2, 32, 16, L_, L_);
  int cur = 0;

  for (int j0 = 0; j0 < L_; j0 += 32) {
    v16h kt0 = load_a_tile_f16(kh, E_, b * L_ + j0,      h * D_, lane);
    v16h kt1 = load_a_tile_f16(kh, E_, b * L_ + j0 + 16, h * D_, lane);
    v8f s0 = {}, s1 = {};
    s0 = WMMA_F16(qa, kt0, s0);
    s1 = WMMA_F16(qa, kt1, s1);

    // issue DMA for next bias tile, then wait for the current one
    if (j0 + 32 < L_) {
      tdm_load_2d(biasLds[cur ^ 1], biasbh + (size_t)row0 * L_ + j0 + 32, 2, 32, 16, L_, L_);
      asm volatile("s_wait_tensorcnt 0x1" ::: "memory");
    } else {
      asm volatile("s_wait_tensorcnt 0x0" ::: "memory");
    }
    const float* bt = biasLds[cur];
    cur ^= 1;

    float kbias0 = (1.0f - mrow[j0 + nloc]) * (-10000.0f);
    float kbias1 = (1.0f - mrow[j0 + 16 + nloc]) * (-10000.0f);

    float sa0[8], sa1[8];
#pragma unroll
    for (int r = 0; r < 8; ++r) {
      sa0[r] = s0[r] * scale + gv * bt[(koff + r) * 32 + nloc]      + kbias0;
      sa1[r] = s1[r] * scale + gv * bt[(koff + r) * 32 + 16 + nloc] + kbias1;
    }

#pragma unroll
    for (int r = 0; r < 8; ++r) {
      float mx = fmaxf(sa0[r], sa1[r]);
#pragma unroll
      for (int m = 8; m >= 1; m >>= 1) mx = fmaxf(mx, __shfl_xor(mx, m, 32));
      float nm = fmaxf(mstate[r], mx);
      float alpha = __expf(mstate[r] - nm);
      float p0 = __expf(sa0[r] - nm);
      float p1 = __expf(sa1[r] - nm);
      float rs = p0 + p1;
#pragma unroll
      for (int m = 8; m >= 1; m >>= 1) rs += __shfl_xor(rs, m, 32);
      lstate[r] = lstate[r] * alpha + rs;
      mstate[r] = nm;
      o0[r] *= alpha;
      o1[r] *= alpha;
      pT[koff + r][nloc]      = (_Float16)p0;
      pT[koff + r][16 + nloc] = (_Float16)p1;
    }
    __syncthreads();

    v16h pa;
#pragma unroll
    for (int i = 0; i < 8; ++i) {
      pa[i]     = pT[nloc][koff + i];
      pa[i + 8] = pT[nloc][16 + koff + i];
    }
    __syncthreads();

    v16h vb0 = load_b_vt(vt, M_, h * D_,      b * L_ + j0, lane);
    v16h vb1 = load_b_vt(vt, M_, h * D_ + 16, b * L_ + j0, lane);
    o0 = WMMA_F16(pa, vb0, o0);
    o1 = WMMA_F16(pa, vb1, o1);
  }

#pragma unroll
  for (int r = 0; r < 8; ++r) {
    int row = row0 + koff + r;
    float inv = 1.0f / fmaxf(lstate[r], 1e-20f);
    float rm = mrow[row];
    size_t base = ((size_t)b * L_ + row) * E_ + h * D_;
    out[base + nloc]      = o0[r] * inv * rm;
    out[base + 16 + nloc] = o1[r] * inv * rm;
  }
}

// ---------------------------------------------------------------------------
// Masked mean pool + classification head ([B, NC] output).
// ---------------------------------------------------------------------------
__global__ __launch_bounds__(256)
void head_kernel(const float* __restrict__ x, const float* __restrict__ mask,
                 const float* __restrict__ g, const float* __restrict__ bta,
                 const float* __restrict__ w1, const float* __restrict__ b1,
                 const float* __restrict__ w2, const float* __restrict__ b2,
                 float* __restrict__ out) {
  __shared__ float sh[E_];
  __shared__ float red[E_];
  int b = blockIdx.x, t = threadIdx.x;
  float s = 0.0f, msum = 0.0f;
  for (int l = 0; l < L_; ++l) {
    float mk = mask[b * L_ + l];
    msum += mk;
    s += x[((size_t)b * L_ + l) * E_ + t] * mk;
  }
  float pooled = s / fmaxf(msum, 1.0f);

  red[t] = pooled;
  __syncthreads();
  for (int off = 128; off > 0; off >>= 1) {
    if (t < off) red[t] += red[t + off];
    __syncthreads();
  }
  float mean = red[0] / (float)E_;
  __syncthreads();
  float d = pooled - mean;
  red[t] = d * d;
  __syncthreads();
  for (int off = 128; off > 0; off >>= 1) {
    if (t < off) red[t] += red[t + off];
    __syncthreads();
  }
  float rstd = rsqrtf(red[0] / (float)E_ + 1e-5f);
  __syncthreads();

  sh[t] = d * rstd * g[t] + bta[t];
  __syncthreads();
  float acc = b1[t];
  for (int k2 = 0; k2 < E_; ++k2) acc += sh[k2] * w1[(size_t)k2 * E_ + t];
  float t1 = gelu_f(acc);
  __syncthreads();
  sh[t] = t1;
  __syncthreads();
  if (t < NC_) {
    float a = b2[t];
    for (int e = 0; e < E_; ++e) a += sh[e] * w2[(size_t)e * NC_ + t];
    out[b * NC_ + t] = a;
  }
}

// ---------------------------------------------------------------------------
extern "C" void kernel_launch(void* const* d_in, const int* in_sizes, int n_in,
                              void* d_out, int out_size, void* d_ws, size_t ws_size,
                              hipStream_t stream) {
  const int*   tokens = (const int*)  d_in[0];
  const float* xpair  = (const float*)d_in[1];
  const float* bias   = (const float*)d_in[2];
  const float* mask   = (const float*)d_in[3];
  const float* temb   = (const float*)d_in[4];
  const float* pe     = (const float*)d_in[5];
  const float* pw1    = (const float*)d_in[6];
  const float* pb1    = (const float*)d_in[7];
  const float* pw2    = (const float*)d_in[8];
  const float* pb2    = (const float*)d_in[9];
  const float* ln1g   = (const float*)d_in[10];
  const float* ln1b   = (const float*)d_in[11];
  const float* wq     = (const float*)d_in[12];
  const float* wk     = (const float*)d_in[13];
  const float* wv     = (const float*)d_in[14];
  const float* wo     = (const float*)d_in[15];
  const float* bq     = (const float*)d_in[16];
  const float* bk     = (const float*)d_in[17];
  const float* bvv    = (const float*)d_in[18];
  const float* bo     = (const float*)d_in[19];
  const float* gate   = (const float*)d_in[20];
  const float* ln2g   = (const float*)d_in[21];
  const float* ln2b   = (const float*)d_in[22];
  const float* fw1    = (const float*)d_in[23];
  const float* fb1    = (const float*)d_in[24];
  const float* fw2    = (const float*)d_in[25];
  const float* fb2    = (const float*)d_in[26];
  const float* hlng   = (const float*)d_in[27];
  const float* hlnb   = (const float*)d_in[28];
  const float* hw1    = (const float*)d_in[29];
  const float* hb1    = (const float*)d_in[30];
  const float* hw2    = (const float*)d_in[31];
  const float* hb2    = (const float*)d_in[32];
  float* out = (float*)d_out;

  // f32 region
  float* X    = (float*)d_ws;
  float* XN   = X    + (size_t)M_ * E_;
  float* AO   = XN   + (size_t)M_ * E_;
  float* HIDb = AO   + (size_t)M_ * E_;
  float* PH   = HIDb + (size_t)M_ * HID_;
  // f16 region
  _Float16* FLAT = (_Float16*)(PH + (size_t)M_ * E_);
  _Float16* W1H  = FLAT + (size_t)M_ * KPAD_;
  _Float16* W2H  = W1H  + (size_t)KPAD_ * E_;
  _Float16* WQH  = W2H  + (size_t)E_ * E_;
  _Float16* WKH  = WQH  + (size_t)NL_ * E_ * E_;
  _Float16* WVH  = WKH  + (size_t)NL_ * E_ * E_;
  _Float16* WOH  = WVH  + (size_t)NL_ * E_ * E_;
  _Float16* FW1H = WOH  + (size_t)NL_ * E_ * E_;
  _Float16* FW2H = FW1H + (size_t)NL_ * E_ * HID_;
  _Float16* QH   = FW2H + (size_t)NL_ * E_ * HID_;
  _Float16* KH   = QH   + (size_t)M_ * E_;
  _Float16* VTH  = KH   + (size_t)M_ * E_;   // [E_][M_]

  const int NEE = NL_ * E_ * E_;
  const int NEH = NL_ * E_ * HID_;
  cvt_w1_kernel<<<(KPAD_ * E_ + 255) / 256, 256, 0, stream>>>(pw1, W1H);
  cvt_f16_kernel<<<(E_ * E_ + 255) / 256, 256, 0, stream>>>(pw2, W2H, E_ * E_);
  cvt_f16_kernel<<<(NEE + 255) / 256, 256, 0, stream>>>(wq, WQH, NEE);
  cvt_f16_kernel<<<(NEE + 255) / 256, 256, 0, stream>>>(wk, WKH, NEE);
  cvt_f16_kernel<<<(NEE + 255) / 256, 256, 0, stream>>>(wv, WVH, NEE);
  cvt_f16_kernel<<<(NEE + 255) / 256, 256, 0, stream>>>(wo, WOH, NEE);
  cvt_f16_kernel<<<(NEH + 255) / 256, 256, 0, stream>>>(fw1, FW1H, NEH);
  cvt_f16_kernel<<<(NEH + 255) / 256, 256, 0, stream>>>(fw2, FW2H, NEH);
  pair_flat_kernel<<<(M_ * KPAD_ + 255) / 256, 256, 0, stream>>>(xpair, FLAT);

  dim3 gblk(128);
  dim3 lblk(32);

  pair_hid_kernel<<<dim3(M_ / 64, E_ / 32), gblk, 0, stream>>>(FLAT, W1H, pb1, PH);
  pair_out_kernel<<<dim3(M_ / 64, E_ / 32), gblk, 0, stream>>>(PH, W2H, pb2, tokens, temb, pe, X);

  for (int i = 0; i < NL_; ++i) {
    layernorm_kernel<<<M_, lblk, 0, stream>>>(X, ln1g + i * E_, ln1b + i * E_, XN);
    gemm_wmma_kernel<<<dim3(M_ / 64, E_ / 32), gblk, 0, stream>>>(
        XN, E_, WQH + (size_t)i * E_ * E_, E_, bq + i * E_, nullptr,
        nullptr, E_, QH, nullptr, E_, 0);
    gemm_wmma_kernel<<<dim3(M_ / 64, E_ / 32), gblk, 0, stream>>>(
        XN, E_, WKH + (size_t)i * E_ * E_, E_, bk + i * E_, nullptr,
        nullptr, E_, KH, nullptr, E_, 0);
    gemm_wmma_kernel<<<dim3(M_ / 64, E_ / 32), gblk, 0, stream>>>(
        XN, E_, WVH + (size_t)i * E_ * E_, E_, bvv + i * E_, nullptr,
        nullptr, E_, nullptr, VTH, E_, 0);
    attention_kernel<<<dim3(L_ / 16, H_, B_), lblk, 0, stream>>>(
        QH, KH, VTH, bias, mask, gate, AO, i);
    gemm_wmma_kernel<<<dim3(M_ / 64, E_ / 32), gblk, 0, stream>>>(
        AO, E_, WOH + (size_t)i * E_ * E_, E_, bo + i * E_, X,
        X, E_, nullptr, nullptr, E_, 0);

    layernorm_kernel<<<M_, lblk, 0, stream>>>(X, ln2g + i * E_, ln2b + i * E_, XN);
    gemm_wmma_kernel<<<dim3(M_ / 64, HID_ / 32), gblk, 0, stream>>>(
        XN, E_, FW1H + (size_t)i * E_ * HID_, HID_, fb1 + i * HID_, nullptr,
        HIDb, HID_, nullptr, nullptr, E_, 2);
    gemm_wmma_kernel<<<dim3(M_ / 64, E_ / 32), gblk, 0, stream>>>(
        HIDb, HID_, FW2H + (size_t)i * HID_ * E_, E_, fb2 + i * E_, X,
        X, E_, nullptr, nullptr, HID_, 0);
  }

  head_kernel<<<B_, 256, 0, stream>>>(X, mask, hlng, hlnb, hw1, hb1, hw2, hb2, out);
  (void)in_sizes; (void)n_in; (void)out_size; (void)ws_size;
}
